// MaxFiltering_73864847557138
// MI455X (gfx1250) — matrix operand order repaired and checked by reading, not compiled, not run on hardware
//
#include <hip/hip_runtime.h>
#include <stdint.h>

#define CCH 256
#define TPB 256

typedef __attribute__((ext_vector_type(16))) __bf16 v16bf;
typedef __attribute__((ext_vector_type(8)))  float  v8f;

union AFrag { uint4 q[2]; v16bf v; };

__device__ __forceinline__ unsigned short f2bf(float f) {
  unsigned u = __float_as_uint(f);
  u += 0x7FFFu + ((u >> 16) & 1u);           // round-to-nearest-even
  return (unsigned short)(u >> 16);
}
__device__ __forceinline__ float bf2f(unsigned short h) {
  return __uint_as_float(((unsigned)h) << 16);
}

// ---------------------------------------------------------------------------
// Pack conv weights (OIHW f32) into exact WMMA B-fragment layout, bf16 hi/lo.
// Index: (((tap*8 + ks)*16 + ntile)*32 + lane)*8 + vgpr   (one dword = 2 bf16)
// B is 32x16 (KxN): lanes 0-15 -> K 0..7,16..23, lanes 16-31 -> K 8..15,24..31; n = lane&15
// ---------------------------------------------------------------------------
__global__ __launch_bounds__(TPB)
void pack_w(const float* __restrict__ w, unsigned* __restrict__ Bhi,
            unsigned* __restrict__ Blo) {
  int idx  = blockIdx.x * TPB + threadIdx.x;        // 0..294911
  int v    = idx & 7;
  int lane = (idx >> 3) & 31;
  int tn   = (idx >> 8) & 15;
  int ks   = (idx >> 12) & 7;
  int tap  = idx >> 15;                             // 0..8
  int n  = tn * 16 + (lane & 15);
  int kl = (v >> 2) * 16 + (lane >> 4) * 8 + (v & 3) * 2;
  int i  = ks * 32 + kl;                            // input channel
  float w0 = w[(n * CCH + i) * 9 + tap];
  float w1 = w[(n * CCH + i + 1) * 9 + tap];
  unsigned short h0 = f2bf(w0), h1 = f2bf(w1);
  unsigned short l0 = f2bf(w0 - bf2f(h0));
  unsigned short l1 = f2bf(w1 - bf2f(h1));
  Bhi[idx] = (unsigned)h0 | ((unsigned)h1 << 16);
  Blo[idx] = (unsigned)l0 | ((unsigned)l1 << 16);
}

// ---------------------------------------------------------------------------
// NCHW f32 -> NHWC bf16 hi/lo (channels contiguous for 16B A-fragment loads)
// ---------------------------------------------------------------------------
template <int H>
__global__ __launch_bounds__(TPB)
void cvt_x(const float* __restrict__ x, unsigned short* __restrict__ xh,
           unsigned short* __restrict__ xl) {
  constexpr int HW  = H * H;
  constexpr int LHW = __builtin_ctz(HW);
  int idx = blockIdx.x * TPB + threadIdx.x;        // NHWC flat index
  int cch = idx & 255;
  int pix = idx >> 8;                              // 0 .. 2*HW-1
  int b   = pix >> LHW;
  int p   = pix & (HW - 1);
  float v = x[((b << 8) + cch) * HW + p];
  unsigned short h = f2bf(v);
  xh[idx] = h;
  xl[idx] = f2bf(v - bf2f(h));
}

// ---------------------------------------------------------------------------
// Implicit-GEMM 3x3 conv, register-blocked: one wave computes a 32(M)x32(N)
// output block (2x2 WMMA tiles, 4 f32 accumulators). K = 9*256, tap-major.
// bf16x3 split precision: hi*hi + hi*lo + lo*hi, f32 accumulate.
// Per K-step: 8 A-loads + 8 B-loads (b128) feed 12 WMMAs.
// ---------------------------------------------------------------------------
template <int H>
__global__ __launch_bounds__(TPB)
void conv_wmma(const unsigned short* __restrict__ xhi,
               const unsigned short* __restrict__ xlo,
               const unsigned* __restrict__ Bhi,
               const unsigned* __restrict__ Blo,
               const float* __restrict__ bias,
               float* __restrict__ y) {
  constexpr int W   = H;
  constexpr int HW  = H * W;
  constexpr int LW  = __builtin_ctz(W);
  constexpr int LHW = __builtin_ctz(HW);

  const int lane = threadIdx.x & 31;
  const int wv   = blockIdx.x * 8 + (threadIdx.x >> 5);
  const int tm2  = wv >> 3;            // 32-row M block  (N2 tiles = 8)
  const int tn2  = wv & 7;             // 32-col N block
  const int half = lane >> 4;
  const int r15  = lane & 15;

  // A-side pixels for this lane's two M tiles
  const int pm0 = tm2 * 32 + r15;
  const int pm1 = pm0 + 16;
  const int bA0 = pm0 >> LHW, pA0 = pm0 & (HW - 1);
  const int hA0 = pA0 >> LW,  wA0 = pA0 & (W - 1);
  const int bA1 = pm1 >> LHW, pA1 = pm1 & (HW - 1);
  const int hA1 = pA1 >> LW,  wA1 = pA1 & (W - 1);
  const int baseC0 = ((bA0 << LHW) + (hA0 << LW) + wA0) * CCH + half * 8;
  const int baseC1 = ((bA1 << LHW) + (hA1 << LW) + wA1) * CCH + half * 8;

  // Output channels for this lane's two N tiles
  const int n0 = tn2 * 32 + r15;
  const int n1 = n0 + 16;
  const float bv0 = bias[n0];
  const float bv1 = bias[n1];
  v8f c00, c01, c10, c11;
#pragma unroll
  for (int r = 0; r < 8; ++r) { c00[r] = bv0; c01[r] = bv1; c10[r] = bv0; c11[r] = bv1; }

  // Packed-B base pointers for the two N tiles (hi/lo)
  const int tnA = tn2 * 2, tnB = tn2 * 2 + 1;
  const unsigned* __restrict__ bh0 = Bhi + (tnA << 8) + (lane << 3);
  const unsigned* __restrict__ bl0 = Blo + (tnA << 8) + (lane << 3);
  const unsigned* __restrict__ bh1 = Bhi + (tnB << 8) + (lane << 3);
  const unsigned* __restrict__ bl1 = Blo + (tnB << 8) + (lane << 3);

#pragma unroll 1
  for (int tap = 0; tap < 9; ++tap) {
    const int dh = tap / 3 - 1, dw = tap % 3 - 1;
    const int toff = (dh * W + dw) * CCH;
    const bool v0 = ((unsigned)(hA0 + dh) < (unsigned)H) && ((unsigned)(wA0 + dw) < (unsigned)W);
    const bool v1 = ((unsigned)(hA1 + dh) < (unsigned)H) && ((unsigned)(wA1 + dw) < (unsigned)W);
    const unsigned short* __restrict__ pH0 = xhi + baseC0 + toff;
    const unsigned short* __restrict__ pL0 = xlo + baseC0 + toff;
    const unsigned short* __restrict__ pH1 = xhi + baseC1 + toff;
    const unsigned short* __restrict__ pL1 = xlo + baseC1 + toff;

#pragma unroll
    for (int ks = 0; ks < 8; ++ks) {
      const int i0 = ks << 5;
      AFrag a0h, a0l, a1h, a1l;
      if (v0) {
        a0h.q[0] = *(const uint4*)(pH0 + i0);
        a0h.q[1] = *(const uint4*)(pH0 + i0 + 16);
        a0l.q[0] = *(const uint4*)(pL0 + i0);
        a0l.q[1] = *(const uint4*)(pL0 + i0 + 16);
      } else {
        a0h.q[0] = make_uint4(0u,0u,0u,0u); a0h.q[1] = make_uint4(0u,0u,0u,0u);
        a0l.q[0] = make_uint4(0u,0u,0u,0u); a0l.q[1] = make_uint4(0u,0u,0u,0u);
      }
      if (v1) {
        a1h.q[0] = *(const uint4*)(pH1 + i0);
        a1h.q[1] = *(const uint4*)(pH1 + i0 + 16);
        a1l.q[0] = *(const uint4*)(pL1 + i0);
        a1l.q[1] = *(const uint4*)(pL1 + i0 + 16);
      } else {
        a1h.q[0] = make_uint4(0u,0u,0u,0u); a1h.q[1] = make_uint4(0u,0u,0u,0u);
        a1l.q[0] = make_uint4(0u,0u,0u,0u); a1l.q[1] = make_uint4(0u,0u,0u,0u);
      }

      const int soff = (tap * 8 + ks) * 4096;     // 16 ntiles * 32 lanes * 8 dw
      AFrag b0h, b0l, b1h, b1l;
      b0h.q[0] = *(const uint4*)(bh0 + soff);
      b0h.q[1] = *(const uint4*)(bh0 + soff + 4);
      b0l.q[0] = *(const uint4*)(bl0 + soff);
      b0l.q[1] = *(const uint4*)(bl0 + soff + 4);
      b1h.q[0] = *(const uint4*)(bh1 + soff);
      b1h.q[1] = *(const uint4*)(bh1 + soff + 4);
      b1l.q[0] = *(const uint4*)(bl1 + soff);
      b1l.q[1] = *(const uint4*)(bl1 + soff + 4);

      // 12 WMMAs: 2x2 tiles x (hi*hi + hi*lo + lo*hi)
      c00 = __builtin_amdgcn_wmma_f32_16x16x32_bf16(false, a0h.v, false, b0h.v, (short)0, c00, false, false);
      c00 = __builtin_amdgcn_wmma_f32_16x16x32_bf16(false, a0h.v, false, b0l.v, (short)0, c00, false, false);
      c00 = __builtin_amdgcn_wmma_f32_16x16x32_bf16(false, a0l.v, false, b0h.v, (short)0, c00, false, false);
      c01 = __builtin_amdgcn_wmma_f32_16x16x32_bf16(false, a0h.v, false, b1h.v, (short)0, c01, false, false);
      c01 = __builtin_amdgcn_wmma_f32_16x16x32_bf16(false, a0h.v, false, b1l.v, (short)0, c01, false, false);
      c01 = __builtin_amdgcn_wmma_f32_16x16x32_bf16(false, a0l.v, false, b1h.v, (short)0, c01, false, false);
      c10 = __builtin_amdgcn_wmma_f32_16x16x32_bf16(false, a1h.v, false, b0h.v, (short)0, c10, false, false);
      c10 = __builtin_amdgcn_wmma_f32_16x16x32_bf16(false, a1h.v, false, b0l.v, (short)0, c10, false, false);
      c10 = __builtin_amdgcn_wmma_f32_16x16x32_bf16(false, a1l.v, false, b0h.v, (short)0, c10, false, false);
      c11 = __builtin_amdgcn_wmma_f32_16x16x32_bf16(false, a1h.v, false, b1h.v, (short)0, c11, false, false);
      c11 = __builtin_amdgcn_wmma_f32_16x16x32_bf16(false, a1h.v, false, b1l.v, (short)0, c11, false, false);
      c11 = __builtin_amdgcn_wmma_f32_16x16x32_bf16(false, a1l.v, false, b1h.v, (short)0, c11, false, false);
    }
  }

  // D layout: lane owns 8 consecutive pixels (M = half*8 + r) of one channel.
  const int mb0 = tm2 * 32 + half * 8;
  const int mb1 = mb0 + 16;
  {
    const int bO = mb0 >> LHW, pO = mb0 & (HW - 1);
    float* d0 = y + (((bO << 8) + n0) << LHW) + pO;
    *(float4*)(d0)     = make_float4(c00[0], c00[1], c00[2], c00[3]);
    *(float4*)(d0 + 4) = make_float4(c00[4], c00[5], c00[6], c00[7]);
    float* d1 = y + (((bO << 8) + n1) << LHW) + pO;
    *(float4*)(d1)     = make_float4(c01[0], c01[1], c01[2], c01[3]);
    *(float4*)(d1 + 4) = make_float4(c01[4], c01[5], c01[6], c01[7]);
  }
  {
    const int bO = mb1 >> LHW, pO = mb1 & (HW - 1);
    float* d0 = y + (((bO << 8) + n0) << LHW) + pO;
    *(float4*)(d0)     = make_float4(c10[0], c10[1], c10[2], c10[3]);
    *(float4*)(d0 + 4) = make_float4(c10[4], c10[5], c10[6], c10[7]);
    float* d1 = y + (((bO << 8) + n1) << LHW) + pO;
    *(float4*)(d1)     = make_float4(c11[0], c11[1], c11[2], c11[3]);
    *(float4*)(d1 + 4) = make_float4(c11[4], c11[5], c11[6], c11[7]);
  }
}

// ---------------------------------------------------------------------------
// 2x down-resize with jax-style antialiased triangle kernel + edge renorm.
// dst is (H,H); src is (2H,2H). NCHW, 512 planes.
// ---------------------------------------------------------------------------
template <int H>
__global__ __launch_bounds__(TPB)
void resize_down2(const float* __restrict__ src, float* __restrict__ dst) {
  constexpr int W = H, H2 = 2 * H, W2 = 2 * W;
  constexpr int LW = __builtin_ctz(W), LH = __builtin_ctz(H);
  int idx = blockIdx.x * TPB + threadIdx.x;
  int w  = idx & (W - 1);
  int t  = idx >> LW;
  int h  = t & (H - 1);
  int cp = t >> LH;
  const float* s = src + (long)cp * H2 * W2;
  const float wt[4] = {0.25f, 0.75f, 0.75f, 0.25f};
  float wy[4], wx[4]; int iy[4], ix[4];
  float sy = 0.f, sx = 0.f;
#pragma unroll
  for (int k = 0; k < 4; ++k) {
    iy[k] = 2 * h - 1 + k;
    wy[k] = ((unsigned)iy[k] < (unsigned)H2) ? wt[k] : 0.f;  sy += wy[k];
    ix[k] = 2 * w - 1 + k;
    wx[k] = ((unsigned)ix[k] < (unsigned)W2) ? wt[k] : 0.f;  sx += wx[k];
  }
  float acc = 0.f;
#pragma unroll
  for (int ky = 0; ky < 4; ++ky) {
    if (wy[ky] == 0.f) continue;
    const float* row = s + (long)iy[ky] * W2;
    float ra = 0.f;
#pragma unroll
    for (int kx = 0; kx < 4; ++kx)
      if (wx[kx] != 0.f) ra += wx[kx] * row[ix[kx]];
    acc += wy[ky] * ra;
  }
  dst[idx] = acc / (sy * sx);
}

// ---------------------------------------------------------------------------
// 2x up-resize, half-pixel bilinear (renormalized edges == clamp for 2 taps).
// dst is (H,H); src is (H/2,H/2).
// ---------------------------------------------------------------------------
template <int H>
__global__ __launch_bounds__(TPB)
void resize_up2(const float* __restrict__ src, float* __restrict__ dst) {
  constexpr int Hd = H, Wd = H, Hs = H / 2, Ws = H / 2;
  constexpr int LWd = __builtin_ctz(Wd), LHd = __builtin_ctz(Hd);
  int idx = blockIdx.x * TPB + threadIdx.x;
  int w  = idx & (Wd - 1);
  int t  = idx >> LWd;
  int h  = t & (Hd - 1);
  int cp = t >> LHd;

  float sfy = 0.5f * h - 0.25f;
  int   y0  = (int)floorf(sfy);
  float fy  = sfy - (float)y0;
  float wy0 = 1.f - fy, wy1 = fy;
  int   y1  = y0 + 1;
  if (y0 < 0)            { y0 = 0; y1 = 0; wy0 = 1.f; wy1 = 0.f; }
  else if (y1 > Hs - 1)  { y1 = y0;        wy0 = 1.f; wy1 = 0.f; }

  float sfx = 0.5f * w - 0.25f;
  int   x0  = (int)floorf(sfx);
  float fx  = sfx - (float)x0;
  float wx0 = 1.f - fx, wx1 = fx;
  int   x1  = x0 + 1;
  if (x0 < 0)            { x0 = 0; x1 = 0; wx0 = 1.f; wx1 = 0.f; }
  else if (x1 > Ws - 1)  { x1 = x0;        wx0 = 1.f; wx1 = 0.f; }

  const float* s = src + (long)cp * Hs * Ws;
  float val = wy0 * (wx0 * s[y0 * Ws + x0] + wx1 * s[y0 * Ws + x1]) +
              wy1 * (wx0 * s[y1 * Ws + x0] + wx1 * s[y1 * Ws + x1]);
  dst[idx] = val;
}

// ---------------------------------------------------------------------------
// Fused: max over sources of 3x3 maxpool (-inf pad) + residual x. -> out
// ---------------------------------------------------------------------------
template <int H>
__device__ __forceinline__ float max3x3(const float* __restrict__ p, int pb,
                                        int h, int w) {
  constexpr int W = H;
  float m = -3.402823e38f;
#pragma unroll
  for (int dh = -1; dh <= 1; ++dh) {
    int hh = h + dh;
    if ((unsigned)hh >= (unsigned)H) continue;
#pragma unroll
    for (int dw = -1; dw <= 1; ++dw) {
      int ww = w + dw;
      if ((unsigned)ww >= (unsigned)W) continue;
      m = fmaxf(m, p[pb + hh * W + ww]);
    }
  }
  return m;
}

template <int H>
__global__ __launch_bounds__(TPB)
void pool_combine(const float* __restrict__ fnat, const float* __restrict__ rdn,
                  const float* __restrict__ rup, const float* __restrict__ x,
                  float* __restrict__ out) {
  constexpr int W = H;
  constexpr int LW = __builtin_ctz(W), LH = __builtin_ctz(H);
  int idx = blockIdx.x * TPB + threadIdx.x;
  int w  = idx & (W - 1);
  int t  = idx >> LW;
  int h  = t & (H - 1);
  int cp = t >> LH;
  const int pb = cp * H * W;
  float m = max3x3<H>(fnat, pb, h, w);
  if (rdn) m = fmaxf(m, max3x3<H>(rdn, pb, h, w));
  if (rup) m = fmaxf(m, max3x3<H>(rup, pb, h, w));
  out[idx] = m + x[idx];
}

// ---------------------------------------------------------------------------
// GroupNorm stats: one block per (batch, group), deterministic tree reduce.
// ---------------------------------------------------------------------------
__global__ __launch_bounds__(TPB)
void gn_stats(const float* __restrict__ z, float2* __restrict__ stats, int HW) {
  __shared__ double ss[TPB];
  __shared__ double sq[TPB];
  const int bg = blockIdx.x;          // 0..63 : b*32 + g
  const int b  = bg >> 5;
  const int g  = bg & 31;
  const float* p = z + ((long)b * CCH + g * 8) * HW;
  const long N = 8L * HW;
  double s = 0.0, q = 0.0;
  for (long i = threadIdx.x; i < N; i += TPB) {
    double v = (double)p[i];
    s += v; q += v * v;
  }
  int tid = threadIdx.x;
  ss[tid] = s; sq[tid] = q;
  __syncthreads();
  for (int st = TPB / 2; st > 0; st >>= 1) {
    if (tid < st) { ss[tid] += ss[tid + st]; sq[tid] += sq[tid + st]; }
    __syncthreads();
  }
  if (tid == 0) {
    double Nd   = (double)N;
    double mean = ss[0] / Nd;
    double var  = sq[0] / Nd - mean * mean;
    float rstd  = (float)(1.0 / sqrt(var + 1e-5));
    stats[bg] = make_float2((float)mean, rstd);
  }
}

template <int H>
__global__ __launch_bounds__(TPB)
void gn_apply(float* __restrict__ z, const float2* __restrict__ stats,
              const float* __restrict__ gamma, const float* __restrict__ beta) {
  constexpr int HW  = H * H;
  constexpr int LHW = __builtin_ctz(HW);
  int idx = blockIdx.x * TPB + threadIdx.x;
  int cp  = idx >> LHW;
  int c   = cp & 255;
  int b   = cp >> 8;
  float2 st = stats[(b << 5) | (c >> 3)];
  float v = (z[idx] - st.x) * st.y * gamma[c] + beta[c];
  z[idx] = v > 0.f ? v : 0.f;
}

// ---------------------------------------------------------------------------
// Host-side per-level dispatch (all spatial sizes are compile-time constants)
// ---------------------------------------------------------------------------
struct Ctx {
  const float* x[5];
  const float* conv_w; const float* conv_b;
  const float* gamma;  const float* beta;
  float* out;
  unsigned short* xhi; unsigned short* xlo;
  float* feats; float* rup; float* rdn;
  unsigned* Bhi; unsigned* Blo;
  float2* stats;
  long off[5];
  hipStream_t stream;
};

template <int H>
static void run_cvt(const Ctx& c, int l) {
  constexpr int E = 2 * CCH * H * H;
  cvt_x<H><<<E / TPB, TPB, 0, c.stream>>>(c.x[l], c.xhi + c.off[l], c.xlo + c.off[l]);
}

template <int H>
static void run_conv(const Ctx& c, int l) {
  constexpr int blocks = (H * H) / 16;      // (2*HW/32 M2-tiles * 8 N2) / 8 waves
  conv_wmma<H><<<blocks, TPB, 0, c.stream>>>(c.xhi + c.off[l], c.xlo + c.off[l],
                                             c.Bhi, c.Blo, c.conv_b, c.feats + c.off[l]);
}

template <int H>
static void run_tail(const Ctx& c, int l) {
  constexpr int E = 2 * CCH * H * H;
  const float* fn = c.feats + c.off[l];
  const float* rd = nullptr;
  const float* ru = nullptr;
  if (l > 0) {   // feats[l-1] is 2H -> downsample to H
    resize_down2<H><<<E / TPB, TPB, 0, c.stream>>>(c.feats + c.off[l - 1], c.rdn);
    rd = c.rdn;
  }
  if (l < 4) {   // feats[l+1] is H/2 -> upsample to H
    resize_up2<H><<<E / TPB, TPB, 0, c.stream>>>(c.feats + c.off[l + 1], c.rup);
    ru = c.rup;
  }
  pool_combine<H><<<E / TPB, TPB, 0, c.stream>>>(fn, rd, ru, c.x[l], c.out + c.off[l]);
  gn_stats<<<64, TPB, 0, c.stream>>>(c.out + c.off[l], c.stats + l * 64, H * H);
  gn_apply<H><<<E / TPB, TPB, 0, c.stream>>>(c.out + c.off[l], c.stats + l * 64,
                                             c.gamma, c.beta);
}

extern "C" void kernel_launch(void* const* d_in, const int* in_sizes, int n_in,
                              void* d_out, int out_size, void* d_ws, size_t ws_size,
                              hipStream_t stream) {
  (void)in_sizes; (void)n_in; (void)out_size; (void)ws_size;
  Ctx c;
  for (int i = 0; i < 5; ++i) c.x[i] = (const float*)d_in[i];
  c.conv_w = (const float*)d_in[5];
  c.conv_b = (const float*)d_in[6];
  c.gamma  = (const float*)d_in[7];
  c.beta   = (const float*)d_in[8];
  c.out    = (float*)d_out;
  c.stream = stream;

  static const int Hs[5] = {128, 64, 32, 16, 8};
  long tot = 0;
  int elems[5];
  for (int l = 0; l < 5; ++l) {
    elems[l] = 2 * CCH * Hs[l] * Hs[l];
    c.off[l] = tot;
    tot     += elems[l];
  }

  char* base = (char*)d_ws;
  size_t cur = 0;
  auto alloc = [&](size_t bytes) -> char* {
    char* p = base + cur;
    cur = (cur + bytes + 255) & ~(size_t)255;
    return p;
  };
  c.xhi   = (unsigned short*)alloc((size_t)tot * 2);
  c.xlo   = (unsigned short*)alloc((size_t)tot * 2);
  c.feats = (float*)alloc((size_t)tot * 4);
  c.rup   = (float*)alloc((size_t)elems[0] * 4);   // largest up-resize target
  c.rdn   = (float*)alloc((size_t)elems[1] * 4);   // largest down-resize target
  c.Bhi   = (unsigned*)alloc(294912u * 4);
  c.Blo   = (unsigned*)alloc(294912u * 4);
  c.stats = (float2*)alloc(5 * 64 * sizeof(float2));

  // 1) pack weights into B-fragment layout (bf16 hi/lo)
  pack_w<<<294912 / TPB, TPB, 0, stream>>>(c.conv_w, c.Bhi, c.Blo);

  // 2) activations -> NHWC bf16 hi/lo
  for (int l = 0; l < 5; ++l) {
    switch (Hs[l]) {
      case 128: run_cvt<128>(c, l); break;
      case  64: run_cvt< 64>(c, l); break;
      case  32: run_cvt< 32>(c, l); break;
      case  16: run_cvt< 16>(c, l); break;
      default:  run_cvt<  8>(c, l); break;
    }
  }

  // 3) WMMA implicit-GEMM convs (register-blocked 32x32 per wave)
  for (int l = 0; l < 5; ++l) {
    switch (Hs[l]) {
      case 128: run_conv<128>(c, l); break;
      case  64: run_conv< 64>(c, l); break;
      case  32: run_conv< 32>(c, l); break;
      case  16: run_conv< 16>(c, l); break;
      default:  run_conv<  8>(c, l); break;
    }
  }

  // 4) per level: resize neighbors, fused pool/max/add, groupnorm, relu
  for (int l = 0; l < 5; ++l) {
    switch (Hs[l]) {
      case 128: run_tail<128>(c, l); break;
      case  64: run_tail< 64>(c, l); break;
      case  32: run_tail< 32>(c, l); break;
      case  16: run_tail< 16>(c, l); break;
      default:  run_tail<  8>(c, l); break;
    }
  }
}
